// Attention_43946105373274
// MI455X (gfx1250) — matrix operand order repaired and checked by reading, not compiled, not run on hardware
//
#include <hip/hip_runtime.h>
#include <hip/hip_bf16.h>

// Problem constants
#define B_   2
#define N_   2048
#define DIM_ 1024
#define H_   16
#define DH_  64
#define ROWS (B_ * N_)        // 4096
#define QKVC (3 * DIM_)       // 3072
#define SCALE_ 0.125f         // DH^-0.5

typedef __attribute__((ext_vector_type(16))) _Float16 v16h;
typedef __attribute__((ext_vector_type(8)))  _Float16 v8h;
typedef __attribute__((ext_vector_type(8)))  float    v8f;

union FragU { v16h v; v8h h[2]; };

__device__ __forceinline__ v8f wmma_f16(v16h a, v16h b, v8f c) {
  return __builtin_amdgcn_wmma_f32_16x16x32_f16(
      /*neg_a=*/false, a, /*neg_b=*/false, b,
      /*c_mod=*/(short)0, c, /*reuse_a=*/false, /*reuse_b=*/false);
}

// ---- CDNA5 async copy: global -> LDS, 16 bytes per lane, tracked by ASYNCcnt ----
__device__ __forceinline__ void async_ld_b128(unsigned lds_addr, const void* g) {
  asm volatile("global_load_async_to_lds_b128 %0, %1, off"
               :: "v"(lds_addr), "v"((unsigned long long)(size_t)g)
               : "memory");
}
__device__ __forceinline__ void wait_async_le1() {
  asm volatile("s_wait_asynccnt 0x1" ::: "memory");
}
__device__ __forceinline__ void wait_async_0() {
  asm volatile("s_wait_asynccnt 0x0" ::: "memory");
}
// Acquire-side workgroup split barrier WITHOUT the implicit vmem fence of
// __syncthreads(). Ordering: each wave executes s_wait_asynccnt (asm volatile,
// cannot reorder with this asm) before signaling, so barrier completion implies
// all waves' async DMAs into the current buffer are done.
__device__ __forceinline__ void wg_barrier() {
  asm volatile("s_barrier_signal -1\n\ts_barrier_wait -1" ::: "memory");
}
// Release-side barrier: architecturally drain this wave's LDS reads (DScnt)
// BEFORE signaling, so no wave can start the DMA that overwrites the buffer
// while reads are in flight. (The compiler may sink the WMMA consumers past a
// plain barrier since they are register-only ops — seen in round-3 asm.)
__device__ __forceinline__ void wg_barrier_release_lds() {
  asm volatile("s_wait_dscnt 0x0\n\ts_barrier_signal -1\n\ts_barrier_wait -1" ::: "memory");
}

// A fragment, 16x32 f16, row-major source (ld = element row stride).
// Layout used here (contiguous k): lanes 0-15 -> row=lane, K 0..15;
//             lanes 16-31 -> row=lane-16, K 16..31.
__device__ __forceinline__ v16h load_frag_a_rm(const _Float16* tile, int ld, int lane) {
  int r  = lane & 15;
  int ko = (lane & 16) ? 16 : 0;
  const _Float16* p = tile + r * ld + ko;
  FragU u;
  u.h[0] = *(const v8h*)(p);
  u.h[1] = *(const v8h*)(p + 8);
  return u.v;
}

// B fragment, 32x16 f16, from "column-contiguous" storage: src points at
// (n=0,k=0) of a tile stored as [N][K] row-major (i.e. logical B transposed).
// Layout used here (contiguous k): lanes 0-15 -> col=lane, K=0..15; lanes 16-31 -> col=lane-16, K=16..31.
__device__ __forceinline__ v16h load_frag_b_ct(const _Float16* tile, int ld, int lane) {
  int n  = lane & 15;
  int ko = (lane & 16) ? 16 : 0;
  const _Float16* p = tile + n * ld + ko;
  FragU u;
  u.h[0] = *(const v8h*)(p);
  u.h[1] = *(const v8h*)(p + 8);
  return u.v;
}

// ---------------- pack / convert kernels ----------------
__global__ __launch_bounds__(256) void cvt_f32_to_f16(const float* __restrict__ in,
                                                      _Float16* __restrict__ out, int n) {
  int i = blockIdx.x * 256 + threadIdx.x;
  if (i < n) out[i] = (_Float16)in[i];
}

// in: R x C row-major f32; out: C x R f16 (out[c*R+r] = in[r*C+c])
__global__ __launch_bounds__(256) void transpose_cvt(const float* __restrict__ in,
                                                     _Float16* __restrict__ out,
                                                     int R, int C) {
  int i = blockIdx.x * 256 + threadIdx.x;
  if (i >= R * C) return;
  int c = i % C, r = i / C;
  out[c * R + r] = (_Float16)in[(size_t)r * C + c];
}

// ---------------- WMMA GEMM: C[MxN] = A[MxK](f16) * Bt[NxK](f16) (+bias) ----------------
// Block: 8 waves, 256 rows x 64 cols of C. Each wave: 32x64 (2Mx4N accumulators).
// B tile (64x32 f16 = 4KB) staged in LDS via async DMA, double-buffered; A fragments
// software-pipelined in registers; fence-free split barriers keep the A prefetch in
// flight across the WMMA block, with an explicit DScnt drain on the release side.
__global__ __launch_bounds__(256) void gemm_wmma_f16(const _Float16* __restrict__ A,
                                                     const _Float16* __restrict__ Bt,
                                                     float* __restrict__ C,
                                                     const float* __restrict__ bias,
                                                     int M, int N, int K) {
  __shared__ _Float16 bstage[2][64 * 32];   // [buf][n_local*32 + k], 8 KB
  int tid  = threadIdx.x;
  int wave = tid >> 5;
  int lane = tid & 31;
  int tileM  = blockIdx.y * 256 + wave * 32;
  int tileN0 = blockIdx.x * 64;

  // async-copy assignment: thread copies 16B; row n_local = tid>>2, chunk = tid&3
  int nloc = tid >> 2;
  int kchk = tid & 3;
  const char* gB = (const char*)(Bt + (size_t)(tileN0 + nloc) * K) + kchk * 16;
  unsigned lds0 = (unsigned)(size_t)(&bstage[0][0]) + (unsigned)tid * 16u;
  unsigned lds1 = (unsigned)(size_t)(&bstage[1][0]) + (unsigned)tid * 16u;

  v8f acc[2][4];
#pragma unroll
  for (int mi = 0; mi < 2; mi++)
#pragma unroll
    for (int c = 0; c < 4; c++) acc[mi][c] = (v8f){};

  // prologue: DMA first B tile, load first A fragments
  async_ld_b128(lds0, gB);
  v16h a0 = load_frag_a_rm(A + (size_t)tileM * K, K, lane);
  v16h a1 = load_frag_a_rm(A + (size_t)(tileM + 16) * K, K, lane);

  int steps = K >> 5;
#pragma unroll 2
  for (int i = 0; i < steps; i++) {
    int buf = i & 1;
    bool more = (i + 1) < steps;
    if (more)
      async_ld_b128(buf ? lds0 : lds1, gB + (size_t)(i + 1) * 64);  // next k-step (32 f16 = 64B)
    v16h na0 = a0, na1 = a1;
    if (more) {
      na0 = load_frag_a_rm(A + (size_t)tileM * K + (i + 1) * 32, K, lane);
      na1 = load_frag_a_rm(A + (size_t)(tileM + 16) * K + (i + 1) * 32, K, lane);
    }
    if (more) wait_async_le1(); else wait_async_0();
    wg_barrier();                          // all waves' DMA for buf complete
    const _Float16* bb = &bstage[buf][0];
    int ko = (lane & 16) ? 16 : 0;
    int cl = lane & 15;
    v16h bf[4];
#pragma unroll
    for (int c = 0; c < 4; c++) {
      const _Float16* p = bb + (16 * c + cl) * 32 + ko;
      FragU u;
      u.h[0] = *(const v8h*)(p);
      u.h[1] = *(const v8h*)(p + 8);
      bf[c] = u.v;
    }
#pragma unroll
    for (int c = 0; c < 4; c++) {
      acc[0][c] = wmma_f16(a0, bf[c], acc[0][c]);
      acc[1][c] = wmma_f16(a1, bf[c], acc[1][c]);
    }
    wg_barrier_release_lds();              // DScnt drained, then signal: buf safe to overwrite
    a0 = na0; a1 = na1;
  }

  int half = lane >> 4, cl = lane & 15;
#pragma unroll
  for (int mi = 0; mi < 2; mi++)
#pragma unroll
    for (int c = 0; c < 4; c++) {
      int coln = tileN0 + 16 * c + cl;
      float bv = bias ? bias[coln] : 0.0f;
#pragma unroll
      for (int r = 0; r < 8; r++)
        C[(size_t)(tileM + 16 * mi + r + 8 * half) * N + coln] = acc[mi][c][r] + bv;
    }
}

// ---------------- rotary + head split ----------------
// qkv f32 [ROWS x 3072]; writes q*SCALE, k as [B,H,N,DH] f16; v as [B,H,DH,N] f16.
__global__ __launch_bounds__(256) void rotary_split(const float* __restrict__ qkv,
                                                    const float* __restrict__ rope,
                                                    _Float16* __restrict__ qb,
                                                    _Float16* __restrict__ kb,
                                                    _Float16* __restrict__ vT) {
  int idx = blockIdx.x * 256 + threadIdx.x;     // B*N*H*DH = 4194304
  int d = idx & 63;
  int h = (idx >> 6) & 15;
  int n = (idx >> 10) & 2047;
  int b = idx >> 21;
  const float* row = qkv + (size_t)(b * N_ + n) * QKVC;
  float f = rope[n * DH_ + d];
  float cf = __cosf(f), sf = __sinf(f);
  int col = h * DH_ + d;
  int pcol = h * DH_ + (d ^ 1);
  // rotate_half: rh[2i] = -x[2i+1], rh[2i+1] = x[2i]
  float sgn = (d & 1) ? 1.0f : -1.0f;
  float q0 = row[col],            qp = row[pcol];
  float k0 = row[DIM_ + col],     kp = row[DIM_ + pcol];
  float v0 = row[2 * DIM_ + col], vp = row[2 * DIM_ + pcol];
  float qr = q0 * cf + sgn * qp * sf;
  float kr = k0 * cf + sgn * kp * sf;
  float vr = v0 * cf + sgn * vp * sf;
  size_t head = (size_t)(b * H_ + h);
  qb[(head * N_ + n) * DH_ + d] = (_Float16)(qr * SCALE_);
  kb[(head * N_ + n) * DH_ + d] = (_Float16)kr;
  vT[(head * DH_ + d) * (size_t)N_ + n] = (_Float16)vr;
}

// ---------------- flash attention (causal, online softmax) ----------------
// grid: (N/128, B*H); 8 waves/block, each wave owns one 16-row query tile.
// Per-wave LDS staging only — wave-internal DS ordering (in-order + explicit
// s_wait_dscnt) is sufficient, no cross-wave barriers needed.
__global__ __launch_bounds__(256) void flash_attn(const _Float16* __restrict__ qb,
                                                  const _Float16* __restrict__ kb,
                                                  const _Float16* __restrict__ vT,
                                                  _Float16* __restrict__ attnb) {
  __shared__ _Float16 plds[8 * 16 * 32];   // per-wave 16x32 P staging (8 KB)
  int wave = threadIdx.x >> 5;
  int lane = threadIdx.x & 31;
  int half = lane >> 4, cl = lane & 15;
  int bh = blockIdx.y;
  int b = bh >> 4, h = bh & 15;
  int qt = blockIdx.x * 128 + wave * 16;

  const _Float16* Q  = qb + (size_t)bh * N_ * DH_;
  const _Float16* Kh = kb + (size_t)bh * N_ * DH_;
  const _Float16* Vt = vT + (size_t)bh * DH_ * N_;
  _Float16* pl = plds + wave * (16 * 32);

  float m[8], l[8];
  v8f o[4];
#pragma unroll
  for (int r = 0; r < 8; r++) { m[r] = -3.0e38f; l[r] = 0.0f; }
#pragma unroll
  for (int c = 0; c < 4; c++) o[c] = (v8f){};

  // q fragments for K=0..31 and K=32..63, reused over all key chunks
  v16h qa0 = load_frag_a_rm(Q + (size_t)qt * DH_ + 0,  DH_, lane);
  v16h qa1 = load_frag_a_rm(Q + (size_t)qt * DH_ + 32, DH_, lane);

  int nch = (qt + 47) >> 5;               // key chunks of 32, covering keys <= qt+15
  for (int c0 = 0; c0 < nch; c0++) {
    int jt = c0 << 5;
    float st[2][8];
#pragma unroll
    for (int t = 0; t < 2; t++) {
      int jb = jt + 16 * t;
      v8f s = (v8f){};
      v16h kf0 = load_frag_b_ct(Kh + (size_t)jb * DH_ + 0,  DH_, lane);
      s = wmma_f16(qa0, kf0, s);
      v16h kf1 = load_frag_b_ct(Kh + (size_t)jb * DH_ + 32, DH_, lane);
      s = wmma_f16(qa1, kf1, s);
      int coln = jb + cl;
#pragma unroll
      for (int r = 0; r < 8; r++) {
        float sv = s[r];
        int rown = qt + r + 8 * half;
        if (coln > rown) sv = -3.0e38f;   // causal mask
        st[t][r] = sv;
      }
    }
    // row max over the 32-key chunk (reduce across the 16 lanes of each half)
    float cmax[8], sc[8], rs[8];
#pragma unroll
    for (int r = 0; r < 8; r++) cmax[r] = fmaxf(st[0][r], st[1][r]);
#pragma unroll
    for (int off = 8; off >= 1; off >>= 1)
#pragma unroll
      for (int r = 0; r < 8; r++)
        cmax[r] = fmaxf(cmax[r], __shfl_xor(cmax[r], off, 32));
#pragma unroll
    for (int r = 0; r < 8; r++) {
      float mn = fmaxf(m[r], cmax[r]);
      sc[r] = __expf(m[r] - mn);
      m[r] = mn;
      st[0][r] = __expf(st[0][r] - mn);
      st[1][r] = __expf(st[1][r] - mn);
      rs[r] = st[0][r] + st[1][r];
    }
#pragma unroll
    for (int off = 8; off >= 1; off >>= 1)
#pragma unroll
      for (int r = 0; r < 8; r++)
        rs[r] += __shfl_xor(rs[r], off, 32);
#pragma unroll
    for (int r = 0; r < 8; r++) {
      l[r] = l[r] * sc[r] + rs[r];
#pragma unroll
      for (int c = 0; c < 4; c++) o[c][r] = o[c][r] * sc[r];
    }
    // reshape P (C-layout) -> A-fragment via per-wave LDS
#pragma unroll
    for (int t = 0; t < 2; t++)
#pragma unroll
      for (int r = 0; r < 8; r++)
        pl[(r + 8 * half) * 32 + 16 * t + cl] = (_Float16)st[t][r];
    asm volatile("s_wait_dscnt 0" ::: "memory");
    v16h pa;
    {
      int ko = half ? 16 : 0;
      const _Float16* pp = pl + cl * 32 + ko;
      FragU u;
      u.h[0] = *(const v8h*)(pp);
      u.h[1] = *(const v8h*)(pp + 8);
      pa = u.v;
    }
#pragma unroll
    for (int c = 0; c < 4; c++) {
      v16h vb = load_frag_b_ct(Vt + (size_t)(16 * c) * N_ + jt, N_, lane);
      o[c] = wmma_f16(pa, vb, o[c]);
    }
  }
  // normalize and write to attnb [B*N x H*DH] f16 (rows = b*N+n, cols = h*DH+d)
#pragma unroll
  for (int r = 0; r < 8; r++) {
    float inv = 1.0f / l[r];
#pragma unroll
    for (int c = 0; c < 4; c++) o[c][r] = o[c][r] * inv;
  }
#pragma unroll
  for (int c = 0; c < 4; c++)
#pragma unroll
    for (int r = 0; r < 8; r++) {
      int nrow = qt + r + 8 * half;
      int col  = h * DH_ + 16 * c + cl;
      attnb[(size_t)(b * N_ + nrow) * DIM_ + col] = (_Float16)o[c][r];
    }
}

// ---------------- launcher ----------------
extern "C" void kernel_launch(void* const* d_in, const int* in_sizes, int n_in,
                              void* d_out, int out_size, void* d_ws, size_t ws_size,
                              hipStream_t stream) {
  const float* x     = (const float*)d_in[0];   // [2,2048,1024]
  const float* rope  = (const float*)d_in[1];   // [2048,64]
  const float* w_qkv = (const float*)d_in[2];   // [1024,3072]
  const float* w_out = (const float*)d_in[3];   // [1024,1024]
  const float* b_out = (const float*)d_in[4];   // [1024]
  float* out = (float*)d_out;                   // [2,2048,1024]

  char* ws = (char*)d_ws;
  _Float16* xb    = (_Float16*)ws; ws += (size_t)ROWS * DIM_ * 2;          // 8 MB
  _Float16* wqkvT = (_Float16*)ws; ws += (size_t)QKVC * DIM_ * 2;          // 6 MB
  _Float16* woutT = (_Float16*)ws; ws += (size_t)DIM_ * DIM_ * 2;          // 2 MB
  float*    qkv   = (float*)ws;    ws += (size_t)ROWS * QKVC * 4;          // 48 MB
  _Float16* qb    = (_Float16*)ws; ws += (size_t)B_ * H_ * N_ * DH_ * 2;   // 8 MB
  _Float16* kb    = (_Float16*)ws; ws += (size_t)B_ * H_ * N_ * DH_ * 2;   // 8 MB
  _Float16* vT    = (_Float16*)ws; ws += (size_t)B_ * H_ * N_ * DH_ * 2;   // 8 MB
  _Float16* attnb = (_Float16*)ws; ws += (size_t)ROWS * DIM_ * 2;          // 8 MB

  // pack inputs to f16 (weights transposed for contiguous B fragments)
  cvt_f32_to_f16<<<(ROWS * DIM_) / 256, 256, 0, stream>>>(x, xb, ROWS * DIM_);
  transpose_cvt<<<(DIM_ * QKVC) / 256, 256, 0, stream>>>(w_qkv, wqkvT, DIM_, QKVC);
  transpose_cvt<<<(DIM_ * DIM_) / 256, 256, 0, stream>>>(w_out, woutT, DIM_, DIM_);

  // QKV projection: [4096x3072] = xb [4096x1024] * wqkvT'
  gemm_wmma_f16<<<dim3(QKVC / 64, ROWS / 256), 256, 0, stream>>>(
      xb, wqkvT, qkv, nullptr, ROWS, QKVC, DIM_);

  // rotary + head split (+ q scaling, v transposed)
  rotary_split<<<(B_ * H_ * N_ * DH_) / 256, 256, 0, stream>>>(qkv, rope, qb, kb, vT);

  // causal flash attention
  flash_attn<<<dim3(N_ / 128, B_ * H_), 256, 0, stream>>>(qb, kb, vT, attnb);

  // output projection with bias
  gemm_wmma_f16<<<dim3(DIM_ / 64, ROWS / 256), 256, 0, stream>>>(
      attnb, woutT, out, b_out, ROWS, DIM_, DIM_);
}
